// LSTMBaseline_41618233099063
// MI455X (gfx1250) — matrix operand order repaired and checked by reading, not compiled, not run on hardware
//
#include <hip/hip_runtime.h>
#include <hip/hip_bf16.h>
#include <stdint.h>

// Problem constants
#define B_SZ 2048
#define T_SZ 1024
#define I_SZ 6
#define H_SZ 38
#define G_SZ 152      // 4*H
#define C_SZ 8
#define NP   160      // gate dim padded to 10 tiles of 16
#define NT   10
#define K0P  64       // layer0 K: I+H = 44 -> 64 (2 chunks of 32)
#define KC0  2
#define K1P  96       // layer1 K: 2H = 76 -> 96 (3 chunks of 32)
#define KC1  3

typedef __attribute__((ext_vector_type(16))) __bf16 v16bf;
typedef __attribute__((ext_vector_type(8)))  __bf16 v8bf;
typedef __attribute__((ext_vector_type(8)))  float  v8f;

// ---------------------------------------------------------------------------
// Prep: pack [Wih | Whh] (zero padded to NP x Kpad) into the per-lane bf16
// B-fragment layout for v_wmma_f32_16x16x32_bf16.
// Fragment f = ntile*KC + kc. Per lane, 16 contiguous bf16 (32B) = the v16bf
// operand. B layout (32x16 bf16): lanes 0-15 hold K=0..7,16..23,
// lanes 16-31 hold K=8..15,24..31.  -> k = kc*32 + (elem>>3)*16 + (lane>>4)*8 + (elem&7), n = nt*16+(lane&15)
// ---------------------------------------------------------------------------
__global__ void lstm_pack_weights(const float* __restrict__ Wih0,
                                  const float* __restrict__ Whh0,
                                  const float* __restrict__ Wih1,
                                  const float* __restrict__ Whh1,
                                  __bf16* __restrict__ pack) {
  int idx = blockIdx.x * blockDim.x + threadIdx.x;
  const int L0  = NT * KC0 * 32 * 16;        // 10240 elements
  const int TOT = L0 + NT * KC1 * 32 * 16;   // 25600 elements
  if (idx >= TOT) return;
  float v = 0.f;
  if (idx < L0) {
    int fi = idx >> 9;                 // 512 elems per fragment
    int lane = (idx >> 4) & 31;
    int elem = idx & 15;
    int ntile = fi >> 1, kc = fi & 1;
    int n = ntile * 16 + (lane & 15);
    int k = kc * 32 + ((elem >> 3) << 4) + ((lane >> 4) << 3) + (elem & 7);
    if (n < G_SZ) {
      if (k < I_SZ)              v = Wih0[n * I_SZ + k];
      else if (k < I_SZ + H_SZ)  v = Whh0[n * H_SZ + (k - I_SZ)];
    }
  } else {
    int j = idx - L0;
    int fi = j >> 9;
    int lane = (j >> 4) & 31;
    int elem = j & 15;
    int ntile = fi / 3, kc = fi % 3;
    int n = ntile * 16 + (lane & 15);
    int k = kc * 32 + ((elem >> 3) << 4) + ((lane >> 4) << 3) + (elem & 7);
    if (n < G_SZ) {
      if (k < H_SZ)              v = Wih1[n * H_SZ + k];
      else if (k < 2 * H_SZ)     v = Whh1[n * H_SZ + (k - H_SZ)];
    }
  }
  pack[idx] = (__bf16)v;
}

// ---------------------------------------------------------------------------
// Fast activations: native v_tanh_f32; sigmoid(x) = 0.5 + 0.5*tanh(x/2).
// ---------------------------------------------------------------------------
__device__ __forceinline__ float fast_rcp(float x) {
#if __has_builtin(__builtin_amdgcn_rcpf)
  return __builtin_amdgcn_rcpf(x);
#else
  return 1.f / x;
#endif
}

__device__ __forceinline__ float fast_tanh(float x) {
#if __has_builtin(__builtin_amdgcn_tanhf)
  return __builtin_amdgcn_tanhf(x);
#else
  return 1.f - 2.f * fast_rcp(1.f + __expf(2.f * x));
#endif
}

__device__ __forceinline__ float sigmf(float x) {
  return fmaf(0.5f, fast_tanh(0.5f * x), 0.5f);
}

// ---------------------------------------------------------------------------
// Fused 2-layer LSTM scan. One block = one 16-row batch tile, 4 waves.
// x_t is prefetched one step ahead via async global->LDS copies (ASYNCcnt).
// ---------------------------------------------------------------------------
__global__ __launch_bounds__(128)
void lstm_fused(const float* __restrict__ x,
                const float* __restrict__ bias0g,
                const float* __restrict__ bias1g,
                const float* __restrict__ Wc,
                const float* __restrict__ bcv,
                const __bf16* __restrict__ pack,
                float* __restrict__ out) {
  __shared__ __bf16 xh0[16 * K0P];   // [x_t | h0_prev | 0pad] row-major bf16
  __shared__ __bf16 xh1[16 * K1P];   // [h1_t | h2_prev | 0pad]
  __shared__ float  gbuf[16 * NP];   // pre-activation gates (f32)
  __shared__ float  c0s[16 * H_SZ];
  __shared__ float  c1s[16 * H_SZ];
  __shared__ float  hf2[16 * H_SZ];  // f32 copy of layer1 h (for projection)
  __shared__ float  biasl[2 * NP];
  __shared__ float  xstg[2][16 * I_SZ];  // double-buffered async x staging

  const int tid  = threadIdx.x;
  const int lane = tid & 31;
  // wave id through readfirstlane -> scalar loop bounds, EXEC untouched at WMMA
  const int wave = __builtin_amdgcn_readfirstlane(tid >> 5);
  const int brow = blockIdx.x * 16;

  // init state + zero pads (pad cols of xh* stay zero forever)
  for (int i = tid; i < 16 * K0P; i += 128) xh0[i] = (__bf16)0.f;
  for (int i = tid; i < 16 * K1P; i += 128) xh1[i] = (__bf16)0.f;
  for (int i = tid; i < 16 * H_SZ; i += 128) { c0s[i] = 0.f; c1s[i] = 0.f; }
  for (int i = tid; i < NP; i += 128) {
    biasl[i]      = (i < G_SZ) ? bias0g[i] : 0.f;
    biasl[NP + i] = (i < G_SZ) ? bias1g[i] : 0.f;
  }

  const int mrow = lane & 15;     // A: batch row / D: column-within-tile
  const int hlf  = lane >> 4;
  const __bf16* w0p = pack;
  const __bf16* w1p = pack + NT * KC0 * 32 * 16;

  // x staging assignment (threads 0..95): element tid of xstg; each lane
  // writes AND reads its own element, so a per-wave s_wait_asynccnt suffices.
  const int sb = tid / I_SZ;      // batch row within tile
  const int sk = tid % I_SZ;      // input feature
  const float* xbase = x + (size_t)(brow + sb) * T_SZ * I_SZ + sk;

  if (tid < 16 * I_SZ) {          // issue async copy of x_{t=0}
    unsigned l0 = (unsigned)(uintptr_t)&xstg[0][tid];
    asm volatile("global_load_async_to_lds_b32 %0, %1, off"
                 :: "v"(l0), "v"(xbase) : "memory");
  }
  __syncthreads();

  // elementwise 2-D mapping: row = tid>>3 (16 rows), col = (tid&7)+8*it
  const int erow  = tid >> 3;
  const int ecol0 = tid & 7;

  for (int t = 0; t < T_SZ; ++t) {
    // ---- stage x_t (from async-landed LDS) + prefetch x_{t+1} ----
    if (tid < 16 * I_SZ) {
      asm volatile("s_wait_asynccnt 0x0" ::: "memory");
      float xv = xstg[t & 1][tid];
      xh0[sb * K0P + sk] = (__bf16)xv;
      int t2 = (t + 1 < T_SZ) ? (t + 1) : t;   // clamp at the end
      unsigned ln = (unsigned)(uintptr_t)&xstg[(t + 1) & 1][tid];
      const float* gn = xbase + (size_t)t2 * I_SZ;
      asm volatile("global_load_async_to_lds_b32 %0, %1, off"
                   :: "v"(ln), "v"(gn) : "memory");
    }
    __syncthreads();

    // -------- layer 0 gate GEMM: (16 x K0P) @ (K0P x NP) --------
    v16bf a0[KC0];
#pragma unroll
    for (int kc = 0; kc < KC0; ++kc) {
      v8bf lo = *(const v8bf*)&xh0[mrow * K0P + kc * 32 + hlf * 8];
      v8bf hi = *(const v8bf*)&xh0[mrow * K0P + kc * 32 + 16 + hlf * 8];
#pragma unroll
      for (int e = 0; e < 8; ++e) { a0[kc][e] = lo[e]; a0[kc][8 + e] = hi[e]; }
    }
    for (int nt = wave; nt < NT; nt += 4) {   // scalar (wave-uniform) bound
      v8f acc = {};
#pragma unroll
      for (int kc = 0; kc < KC0; ++kc) {
        v16bf bf = *(const v16bf*)&w0p[((nt * KC0 + kc) * 32 + lane) * 16];
        acc = __builtin_amdgcn_wmma_f32_16x16x32_bf16(false, a0[kc], false, bf,
                                                      (short)0, acc, false, false);
      }
      int n = nt * 16 + mrow;
      float bv = biasl[n];
#pragma unroll
      for (int v = 0; v < 8; ++v)            // D: row = v + hlf*8, col = n
        gbuf[(v + hlf * 8) * NP + n] = acc[v] + bv;
    }
    __syncthreads();

    // -------- layer 0 elementwise --------
#pragma unroll
    for (int it = 0; it < 5; ++it) {
      int j = ecol0 + it * 8;
      if (j < H_SZ) {
        int gb = erow * NP + j;
        float ig = sigmf(gbuf[gb]);
        float fg = sigmf(gbuf[gb + H_SZ]);
        float gg = fast_tanh(gbuf[gb + 2 * H_SZ]);
        float og = sigmf(gbuf[gb + 3 * H_SZ]);
        int ci = erow * H_SZ + j;
        float c = fg * c0s[ci] + ig * gg;
        float h = og * fast_tanh(c);
        c0s[ci] = c;
        xh0[erow * K0P + I_SZ + j] = (__bf16)h;  // recurrent input for t+1
        xh1[erow * K1P + j]        = (__bf16)h;  // layer-1 step input
      }
    }
    __syncthreads();

    // -------- layer 1 gate GEMM: (16 x K1P) @ (K1P x NP) --------
    v16bf a1[KC1];
#pragma unroll
    for (int kc = 0; kc < KC1; ++kc) {
      v8bf lo = *(const v8bf*)&xh1[mrow * K1P + kc * 32 + hlf * 8];
      v8bf hi = *(const v8bf*)&xh1[mrow * K1P + kc * 32 + 16 + hlf * 8];
#pragma unroll
      for (int e = 0; e < 8; ++e) { a1[kc][e] = lo[e]; a1[kc][8 + e] = hi[e]; }
    }
    for (int nt = wave; nt < NT; nt += 4) {
      v8f acc = {};
#pragma unroll
      for (int kc = 0; kc < KC1; ++kc) {
        v16bf bf = *(const v16bf*)&w1p[((nt * KC1 + kc) * 32 + lane) * 16];
        acc = __builtin_amdgcn_wmma_f32_16x16x32_bf16(false, a1[kc], false, bf,
                                                      (short)0, acc, false, false);
      }
      int n = nt * 16 + mrow;
      float bv = biasl[NP + n];
#pragma unroll
      for (int v = 0; v < 8; ++v)
        gbuf[(v + hlf * 8) * NP + n] = acc[v] + bv;
    }
    __syncthreads();

    // -------- layer 1 elementwise --------
#pragma unroll
    for (int it = 0; it < 5; ++it) {
      int j = ecol0 + it * 8;
      if (j < H_SZ) {
        int gb = erow * NP + j;
        float ig = sigmf(gbuf[gb]);
        float fg = sigmf(gbuf[gb + H_SZ]);
        float gg = fast_tanh(gbuf[gb + 2 * H_SZ]);
        float og = sigmf(gbuf[gb + 3 * H_SZ]);
        int ci = erow * H_SZ + j;
        float c = fg * c1s[ci] + ig * gg;
        float h = og * fast_tanh(c);
        c1s[ci] = c;
        xh1[erow * K1P + H_SZ + j] = (__bf16)h;  // recurrent input for t+1
        hf2[ci] = h;                             // f32 copy (last step used)
      }
    }
    __syncthreads();
  }

  // -------- final projection: out = h2_last @ Wc^T + bc --------
  if (tid < 16 * C_SZ) {
    int bb = tid >> 3, ci = tid & 7;
    float s = bcv[ci];
#pragma unroll
    for (int j = 0; j < H_SZ; ++j) s += hf2[bb * H_SZ + j] * Wc[ci * H_SZ + j];
    out[(size_t)(brow + bb) * C_SZ + ci] = s;
  }
  // drain the final (clamped) prefetch before wave exit
  asm volatile("s_wait_asynccnt 0x0" ::: "memory");
}

extern "C" void kernel_launch(void* const* d_in, const int* in_sizes, int n_in,
                              void* d_out, int out_size, void* d_ws, size_t ws_size,
                              hipStream_t stream) {
  const float* x    = (const float*)d_in[0];
  const float* Wih0 = (const float*)d_in[1];
  const float* Whh0 = (const float*)d_in[2];
  const float* b0   = (const float*)d_in[3];
  const float* Wih1 = (const float*)d_in[4];
  const float* Whh1 = (const float*)d_in[5];
  const float* b1   = (const float*)d_in[6];
  const float* Wc   = (const float*)d_in[7];
  const float* bc   = (const float*)d_in[8];
  __bf16* pack = (__bf16*)d_ws;            // 25600 bf16 = 51.2 KB of scratch

  lstm_pack_weights<<<dim3((25600 + 255) / 256), dim3(256), 0, stream>>>(
      Wih0, Whh0, Wih1, Whh1, pack);
  lstm_fused<<<dim3(B_SZ / 16), dim3(128), 0, stream>>>(
      x, b0, b1, Wc, bc, pack, (float*)d_out);
}
